// RGBD_Block_36893769072720
// MI455X (gfx1250) — matrix operand order repaired and checked by reading, not compiled, not run on hardware
//
#include <hip/hip_runtime.h>

// ---------------- constants ----------------
#define Bc 8
#define Hc 64
#define Wc 64
#define Dc 512
#define Tc 4096                    // H*W
#define NHEADc 8
#define HDc 64
#define NEc 10
#define ROWK 32768                 // W*D == H*D (per-batch row length of the (64, 32768) view)
#define NTOT (Bc*Hc*Wc*Dc)         // 16777216
#define MARGINc 0.05f
#define SIGMAc 9.0f
#define DECAYc 0.28768207245178085f
#define INV_SQRT_ROWK 0.005524271728019903f  // 1/sqrt(32768)

typedef __attribute__((ext_vector_type(16))) __bf16 v16bf;
typedef __attribute__((ext_vector_type(8)))  __bf16 v8bf;
typedef __attribute__((ext_vector_type(8)))  float  v8f;

static __device__ inline unsigned short f2bf(float f) {
  unsigned int u = __float_as_uint(f);
  unsigned int r = u + 0x7FFFu + ((u >> 16) & 1u);
  return (unsigned short)(r >> 16);
}
static __device__ inline float bf2f(unsigned short u) {
  return __uint_as_float(((unsigned int)u) << 16);
}
static __device__ inline __bf16 bfbits(unsigned short u) {
  union { unsigned short s; __bf16 b; } x; x.s = u; return x.b;
}

// ---------------- utility kernels ----------------
__global__ void k_zero(float* p, int n) {
  int i = blockIdx.x * blockDim.x + threadIdx.x;
  if (i < n) p[i] = 0.f;
}

__global__ void k_f32_to_bf16(const float* src, unsigned short* dst, int n) {
  int i = blockIdx.x * blockDim.x + threadIdx.x;
  if (i < n) dst[i] = f2bf(src[i]);
}

// depthwise 3x3 conv + bias + residual; writes fp32 and bf16 copies
__global__ void k_dwconv(const float* __restrict__ x, const float* __restrict__ kern,
                         const float* __restrict__ bias, float* __restrict__ xp,
                         unsigned short* __restrict__ xb) {
  int p = blockIdx.x;                 // b*H*W + h*W + w
  int wpos = p % Wc; int hpos = (p / Wc) % Hc; int b = p / (Hc * Wc);
  size_t base = (size_t)p * Dc;
  for (int d = threadIdx.x; d < Dc; d += blockDim.x) {
    float acc = bias[d];
#pragma unroll
    for (int i = 0; i < 3; i++) {
      int hh = hpos + i - 1; if (hh < 0 || hh >= Hc) continue;
#pragma unroll
      for (int j = 0; j < 3; j++) {
        int ww = wpos + j - 1; if (ww < 0 || ww >= Wc) continue;
        acc += x[(((size_t)b * Hc + hh) * Wc + ww) * Dc + d] * kern[(i * 3 + j) * Dc + d];
      }
    }
    float v = x[base + d] + acc;
    xp[base + d] = v;
    xb[base + d] = f2bf(v);
  }
}

// cls = mean over T of xp per (b,d)
__global__ void k_colmean(const float* __restrict__ xp, float* __restrict__ cls) {
  int b = blockIdx.x;
  int d = blockIdx.y * blockDim.x + threadIdx.x;
  const float* p = xp + (size_t)b * Tc * Dc + d;
  float s = 0.f;
  for (int t = 0; t < Tc; t++) s += p[(size_t)t * Dc];
  cls[b * Dc + d] = s * (1.0f / Tc);
}

// Y[r,n] = X[r,:] . Wm[n,:] + bias[n]   (fp32, small rows)
__global__ void k_small_gemm(const float* __restrict__ X, const float* __restrict__ Wm,
                             const float* __restrict__ bias, float* __restrict__ Y,
                             int K, int N) {
  int r = blockIdx.x;
  int n = blockIdx.y * blockDim.x + threadIdx.x;
  const float* xr = X + (size_t)r * K;
  const float* wr = Wm + (size_t)n * K;
  float acc = bias[n];
  for (int k = 0; k < K; k++) acc += xr[k] * wr[k];
  Y[(size_t)r * N + n] = acc;
}

// ---------------- main WMMA GEMM: C[m,n] = sum_k A[m,k]*Wt[n,k] + bias[n] ----------------
// block = 256 threads (8 waves), block tile 128x128, wave tile 32x64
__global__ void k_gemm_wmma(const unsigned short* __restrict__ A,
                            const unsigned short* __restrict__ Wt,
                            const float* __restrict__ bias,
                            unsigned short* __restrict__ Cout,
                            int N, int K) {
  int lane = threadIdx.x & 31, wave = threadIdx.x >> 5;
  int wm = wave >> 1, wn = wave & 1;
  int m_base = blockIdx.x * 128 + wm * 32;
  int n_base = blockIdx.y * 128 + wn * 64;
  int am = lane & 15, aKb = (lane < 16) ? 0 : 16;
  int bn = lane & 15, bKb = (lane < 16) ? 0 : 16;

  v8f acc[2][4];
#pragma unroll
  for (int i = 0; i < 2; i++)
#pragma unroll
    for (int j = 0; j < 4; j++) acc[i][j] = 0.f;

  for (int kk = 0; kk < K; kk += 32) {
    v16bf af[2];
#pragma unroll
    for (int i = 0; i < 2; i++) {
      const unsigned short* pr = A + (size_t)(m_base + i * 16 + am) * K + kk + aKb;
      v8bf a0 = *(const v8bf*)pr;
      v8bf a1 = *(const v8bf*)(pr + 8);
      v16bf t;
#pragma unroll
      for (int e = 0; e < 8; e++) { t[e] = a0[e]; t[8 + e] = a1[e]; }
      af[i] = t;
    }
    v16bf bfr[4];
#pragma unroll
    for (int j = 0; j < 4; j++)
      bfr[j] = *(const v16bf*)(Wt + (size_t)(n_base + j * 16 + bn) * K + kk + bKb);
#pragma unroll
    for (int i = 0; i < 2; i++)
#pragma unroll
      for (int j = 0; j < 4; j++)
        acc[i][j] = __builtin_amdgcn_wmma_f32_16x16x32_bf16(
            false, af[i], false, bfr[j], (short)0, acc[i][j], false, false);
  }
#pragma unroll
  for (int i = 0; i < 2; i++)
#pragma unroll
    for (int j = 0; j < 4; j++) {
      int n = n_base + j * 16 + (lane & 15);
      int m0 = m_base + i * 16 + ((lane < 16) ? 0 : 8);
      float bs = bias[n];
#pragma unroll
      for (int r2 = 0; r2 < 8; r2++)
        Cout[(size_t)(m0 + r2) * N + n] = f2bf(acc[i][j][r2] + bs);
    }
}

// ---------------- per-batch Gram via WMMA: C[b,m,n] += X[b,m,:klen] . Y[b,n,:klen] ----------------
// grid (B, 16 tiles, KCHUNKS), block = 1 wave
__global__ void k_gram_wmma(const unsigned short* __restrict__ X,
                            const unsigned short* __restrict__ Y,
                            float* __restrict__ C) {
  int b = blockIdx.x, tile = blockIdx.y, kc = blockIdx.z;
  int m_base = (tile >> 2) * 16, n_base = (tile & 3) * 16;
  int lane = threadIdx.x & 31;
  int am = lane & 15, aKb = (lane < 16) ? 0 : 16;
  int bn = lane & 15, bKb = (lane < 16) ? 0 : 16;
  const unsigned short* Xb = X + (size_t)b * 64 * ROWK;
  const unsigned short* Yb = Y + (size_t)b * 64 * ROWK;
  v8f acc = 0.f;
  int kk0 = kc * (ROWK / 8);
  for (int kk = kk0; kk < kk0 + ROWK / 8; kk += 32) {
    const unsigned short* pr = Xb + (size_t)(m_base + am) * ROWK + kk + aKb;
    v8bf a0 = *(const v8bf*)pr;
    v8bf a1 = *(const v8bf*)(pr + 8);
    v16bf af;
#pragma unroll
    for (int e = 0; e < 8; e++) { af[e] = a0[e]; af[8 + e] = a1[e]; }
    v16bf bfr = *(const v16bf*)(Yb + (size_t)(n_base + bn) * ROWK + kk + bKb);
    acc = __builtin_amdgcn_wmma_f32_16x16x32_bf16(false, af, false, bfr, (short)0,
                                                  acc, false, false);
  }
  float* Cb = C + (size_t)b * 4096;
  int n = n_base + (lane & 15);
  int m0 = m_base + ((lane < 16) ? 0 : 8);
#pragma unroll
  for (int r2 = 0; r2 < 8; r2++) atomicAdd(&Cb[(m0 + r2) * 64 + n], acc[r2]);
}

// ---------------- MoE attention path ----------------
// scores[row, t] = q . k / 8 ; row = which*64 + b*8 + head
__global__ void k_scores(const float* __restrict__ qf, const unsigned short* __restrict__ kp,
                         float* __restrict__ sc) {
  int t = blockIdx.y * blockDim.x + threadIdx.x;
  int head = blockIdx.x & 7, b = (blockIdx.x >> 3) & 7, which = blockIdx.x >> 6;
  const float* q = qf + (size_t)(which * Bc + b) * Dc + head * HDc;
  const unsigned short* kr = kp + ((size_t)b * Tc + t) * Dc + head * HDc;
  float a = 0.f;
#pragma unroll
  for (int d = 0; d < HDc; d++) a += q[d] * bf2f(kr[d]);
  sc[(size_t)blockIdx.x * Tc + t] = a * 0.125f;
}

__global__ void k_softmax4096(float* __restrict__ sc) {
  __shared__ float red[256];
  float* p = sc + (size_t)blockIdx.x * Tc;
  int tid = threadIdx.x;
  float mx = -1e30f;
  for (int t = tid; t < Tc; t += 256) mx = fmaxf(mx, p[t]);
  red[tid] = mx; __syncthreads();
  for (int s2 = 128; s2 > 0; s2 >>= 1) {
    if (tid < s2) red[tid] = fmaxf(red[tid], red[tid + s2]);
    __syncthreads();
  }
  mx = red[0]; __syncthreads();
  float sm = 0.f;
  for (int t = tid; t < Tc; t += 256) { float e = __expf(p[t] - mx); p[t] = e; sm += e; }
  red[tid] = sm; __syncthreads();
  for (int s2 = 128; s2 > 0; s2 >>= 1) {
    if (tid < s2) red[tid] += red[tid + s2];
    __syncthreads();
  }
  float inv = 1.0f / red[0];
  for (int t = tid; t < Tc; t += 256) p[t] *= inv;
}

// o[which,b, head*64+d] = sum_t attn * v
__global__ void k_attn_o(const float* __restrict__ attn, const unsigned short* __restrict__ vmoe,
                         float* __restrict__ o) {
  int row = blockIdx.x;
  int head = row & 7, b = (row >> 3) & 7, which = row >> 6;
  int d = threadIdx.x;
  const float* a = attn + (size_t)row * Tc;
  const unsigned short* vr = vmoe + (size_t)b * Tc * Dc + head * HDc + d;
  float acc = 0.f;
  for (int t = 0; t < Tc; t++) acc += a[t] * bf2f(vr[(size_t)t * Dc]);
  o[(size_t)(which * Bc + b) * Dc + head * HDc + d] = acc;
}

// routing + gaussian gates, max-normalized; s[row=which*8+b][t]
__global__ void k_router(const float* __restrict__ tw,
                         const float* __restrict__ rt_w, const float* __restrict__ rt_b,
                         const float* __restrict__ gs_w, const float* __restrict__ gs_b,
                         float* __restrict__ s) {
  int r = blockIdx.x;
  __shared__ float lg[NEc];
  __shared__ float gg[2 * NEc];
  __shared__ float red[256];
  __shared__ float cen_s, wdt_s;
  const float* x = tw + (size_t)r * Dc;
  int tid = threadIdx.x;
  if (tid < NEc) {
    float a = rt_b[tid];
    for (int k = 0; k < Dc; k++) a += x[k] * rt_w[tid * Dc + k];
    lg[tid] = a;
  } else if (tid >= 32 && tid < 32 + 2 * NEc) {
    int j = tid - 32;
    float a = gs_b[j];
    for (int k = 0; k < Dc; k++) a += x[k] * gs_w[j * Dc + k];
    gg[j] = a;
  }
  __syncthreads();
  if (tid == 0) {
    int bi = 0; float bv = lg[0];
    for (int e = 1; e < NEc; e++) if (lg[e] > bv) { bv = lg[e]; bi = e; }
    float cen = tanhf(gg[2 * bi]) * MARGINc + (MARGINc + 0.1f * bi);
    cen_s = fminf(fmaxf(cen, 0.f), 1.f);
    float wid = 1.f / (1.f + __expf(-gg[2 * bi + 1]));
    wdt_s = fmaxf(wid, 0.09f) / SIGMAc;
  }
  __syncthreads();
  float c = cen_s, wd = wdt_s;
  float inv2 = 1.f / (2.f * wd * wd);
  float* out = s + (size_t)r * Tc;
  float mx = 0.f;
  for (int t = tid; t < Tc; t += 256) {
    float tn = t * (1.0f / 4095.0f);
    float v = __expf(-(tn - c) * (tn - c) * inv2);
    out[t] = v;
    mx = fmaxf(mx, v);
  }
  red[tid] = mx; __syncthreads();
  for (int s2 = 128; s2 > 0; s2 >>= 1) {
    if (tid < s2) red[tid] = fmaxf(red[tid], red[tid + s2]);
    __syncthreads();
  }
  float im = 1.f / red[0];
  for (int t = tid; t < Tc; t += 256) out[t] *= im;
}

// Xw = bf16(s_wid * xp), Xh = bf16(s_hei * xp)
__global__ void k_scale(const float* __restrict__ xp, const float* __restrict__ s,
                        unsigned short* __restrict__ Xw, unsigned short* __restrict__ Xh) {
  int p = blockIdx.x;                // b*T + t
  int b = p >> 12, t = p & 4095;
  float sw = s[(size_t)b * Tc + t];
  float sh = s[(size_t)(Bc + b) * Tc + t];
  size_t base = (size_t)p * Dc;
  for (int d = threadIdx.x; d < Dc; d += blockDim.x) {
    float v = xp[base + d];
    Xw[base + d] = f2bf(sw * v);
    Xh[base + d] = f2bf(sh * v);
  }
}

// priors + softmax + decay-gating + M = Sy@Sx ; one block (64 threads) per batch
__global__ void k_combine(const float* __restrict__ C4, unsigned short* __restrict__ Mb) {
  __shared__ float Sx[64][64];
  __shared__ float Sy[64][64];
  int b = blockIdx.x, m = threadIdx.x;
  const float* Cw = C4 + (size_t)b * 4096;                 // C_wid
  const float* Ch = C4 + (size_t)(Bc + b) * 4096;          // C_hei
  const float* Cx = C4 + (size_t)(2 * Bc + b) * 4096;      // Xq.Xk^T
  const float* Cy = C4 + (size_t)(3 * Bc + b) * 4096;      // Yq.Yk^T
  float lnD = logf(DECAYc);

  // --- Sx row m ---
  {
    float mx = -1e30f;
    for (int n = 0; n < 64; n++) mx = fmaxf(mx, Cx[m * 64 + n] * INV_SQRT_ROWK);
    float sm = 0.f;
    float dm = Cw[m * 64 + m];
    for (int n = 0; n < 64; n++) {
      float e = __expf(Cx[m * 64 + n] * INV_SQRT_ROWK - mx);
      sm += e;
      Sx[m][n] = e;
    }
    float inv = 1.f / sm;
    for (int n = 0; n < 64; n++) {
      float dn = Cw[n * 64 + n];
      float pr = fabsf(Cw[m * 64 + n]) / fmaxf(sqrtf(dm * dn), 1e-8f);
      Sx[m][n] = Sx[m][n] * inv * __expf(pr * lnD);
    }
  }
  // --- Sy row m ---
  {
    float mx = -1e30f;
    for (int n = 0; n < 64; n++) mx = fmaxf(mx, Cy[m * 64 + n] * INV_SQRT_ROWK);
    float sm = 0.f;
    float dm = Ch[m * 64 + m];
    for (int n = 0; n < 64; n++) {
      float e = __expf(Cy[m * 64 + n] * INV_SQRT_ROWK - mx);
      sm += e;
      Sy[m][n] = e;
    }
    float inv = 1.f / sm;
    for (int n = 0; n < 64; n++) {
      float dn = Ch[n * 64 + n];
      float pr = fabsf(Ch[m * 64 + n]) / fmaxf(sqrtf(dm * dn), 1e-8f);
      Sy[m][n] = Sy[m][n] * inv * __expf(pr * lnD);
    }
  }
  __syncthreads();
  for (int n = 0; n < 64; n++) {
    float acc = 0.f;
    for (int k = 0; k < 64; k++) acc += Sy[m][k] * Sx[k][n];
    Mb[(size_t)b * 4096 + m * 64 + n] = f2bf(acc);
  }
}

// out[b] (64x32768) = M[b] (64x64) @ V[b] (64x32768) via WMMA; wave per 64x16 column chunk
__global__ void k_final_wmma(const unsigned short* __restrict__ Mb,
                             const unsigned short* __restrict__ V,
                             float* __restrict__ out) {
  int b = blockIdx.x;
  int lane = threadIdx.x & 31, wave = threadIdx.x >> 5;
  int n_base = (blockIdx.y * 8 + wave) * 16;
  int am = lane & 15, aKb = (lane < 16) ? 0 : 16;
  int bn = lane & 15, bKb = (lane < 16) ? 0 : 16;
  const unsigned short* Mbb = Mb + (size_t)b * 4096;
  const unsigned short* Vb = V + (size_t)b * 64 * ROWK;
  v8f acc[4];
#pragma unroll
  for (int mt = 0; mt < 4; mt++) acc[mt] = 0.f;
#pragma unroll
  for (int kk = 0; kk < 64; kk += 32) {
    v16bf bfr;
#pragma unroll
    for (int e = 0; e < 16; e++) {
      int k = kk + bKb + e;
      bfr[e] = bfbits(Vb[(size_t)k * ROWK + n_base + bn]);
    }
#pragma unroll
    for (int mt = 0; mt < 4; mt++) {
      const unsigned short* pr = Mbb + (size_t)(mt * 16 + am) * 64 + kk + aKb;
      v8bf a0 = *(const v8bf*)pr;
      v8bf a1 = *(const v8bf*)(pr + 8);
      v16bf af;
#pragma unroll
      for (int e = 0; e < 8; e++) { af[e] = a0[e]; af[8 + e] = a1[e]; }
      acc[mt] = __builtin_amdgcn_wmma_f32_16x16x32_bf16(false, af, false, bfr, (short)0,
                                                        acc[mt], false, false);
    }
  }
#pragma unroll
  for (int mt = 0; mt < 4; mt++) {
    int n = n_base + (lane & 15);
    int m0 = mt * 16 + ((lane < 16) ? 0 : 8);
#pragma unroll
    for (int r2 = 0; r2 < 8; r2++)
      out[(size_t)b * 64 * ROWK + (size_t)(m0 + r2) * ROWK + n] = acc[mt][r2];
  }
}

// ---------------- host ----------------
extern "C" void kernel_launch(void* const* d_in, const int* in_sizes, int n_in,
                              void* d_out, int out_size, void* d_ws, size_t ws_size,
                              hipStream_t stream) {
  const float* x     = (const float*)d_in[0];
  const float* dw_k  = (const float*)d_in[1];
  const float* dw_b  = (const float*)d_in[2];
  const float* wid_w = (const float*)d_in[3];
  const float* wid_b = (const float*)d_in[4];
  const float* hei_w = (const float*)d_in[5];
  const float* hei_b = (const float*)d_in[6];
  const float* in_w  = (const float*)d_in[7];
  const float* in_b  = (const float*)d_in[8];
  const float* out_w = (const float*)d_in[9];
  const float* out_b = (const float*)d_in[10];
  const float* rt_w  = (const float*)d_in[11];
  const float* rt_b  = (const float*)d_in[12];
  const float* gs_w  = (const float*)d_in[13];
  const float* gs_b  = (const float*)d_in[14];
  const float* xq_w  = (const float*)d_in[15];
  const float* xq_b  = (const float*)d_in[16];
  const float* xk_w  = (const float*)d_in[17];
  const float* xk_b  = (const float*)d_in[18];
  const float* yq_w  = (const float*)d_in[19];
  const float* yq_b  = (const float*)d_in[20];
  const float* yk_w  = (const float*)d_in[21];
  const float* yk_b  = (const float*)d_in[22];
  const float* v_w   = (const float*)d_in[23];
  const float* v_b   = (const float*)d_in[24];
  float* dout = (float*)d_out;

  char* wsp = (char*)d_ws;
  auto alloc = [&](size_t bytes) -> void* {
    void* p = (void*)wsp;
    wsp += (bytes + 255) & ~(size_t)255;
    return p;
  };

  float*          xp   = (float*)alloc((size_t)NTOT * 4);
  unsigned short* xb   = (unsigned short*)alloc((size_t)NTOT * 2);
  unsigned short* kp   = (unsigned short*)alloc((size_t)NTOT * 2);  // later aliased as Xw
  unsigned short* vmoe = (unsigned short*)alloc((size_t)NTOT * 2);  // later aliased as Xh
  unsigned short* xqb  = (unsigned short*)alloc((size_t)NTOT * 2);
  unsigned short* xkb  = (unsigned short*)alloc((size_t)NTOT * 2);
  unsigned short* yqb  = (unsigned short*)alloc((size_t)NTOT * 2);
  unsigned short* ykb  = (unsigned short*)alloc((size_t)NTOT * 2);
  unsigned short* vfin = (unsigned short*)alloc((size_t)NTOT * 2);
  unsigned short* wbf  = (unsigned short*)alloc((size_t)7 * Dc * Dc * 2);
  float* cls   = (float*)alloc((size_t)Bc * Dc * 4);
  float* whcls = (float*)alloc((size_t)2 * Bc * Dc * 4);
  float* qf    = (float*)alloc((size_t)2 * Bc * Dc * 4);
  float* attn  = (float*)alloc((size_t)2 * Bc * NHEADc * Tc * 4);
  float* ovec  = (float*)alloc((size_t)2 * Bc * Dc * 4);
  float* tw    = (float*)alloc((size_t)2 * Bc * Dc * 4);
  float* sgate = (float*)alloc((size_t)2 * Bc * Tc * 4);
  float* C4    = (float*)alloc((size_t)4 * Bc * 4096 * 4);
  unsigned short* Mb = (unsigned short*)alloc((size_t)Bc * 4096 * 2);

  const int WSZ = Dc * Dc;  // 262144
  unsigned short* wbf_k  = wbf + 0 * WSZ;
  unsigned short* wbf_v  = wbf + 1 * WSZ;
  unsigned short* wbf_xq = wbf + 2 * WSZ;
  unsigned short* wbf_xk = wbf + 3 * WSZ;
  unsigned short* wbf_yq = wbf + 4 * WSZ;
  unsigned short* wbf_yk = wbf + 5 * WSZ;
  unsigned short* wbf_vf = wbf + 6 * WSZ;

  // 0) zero Gram accumulators
  k_zero<<<dim3((4 * Bc * 4096 + 255) / 256), 256, 0, stream>>>(C4, 4 * Bc * 4096);

  // 1) depthwise conv + residual
  k_dwconv<<<dim3(Bc * Hc * Wc), 256, 0, stream>>>(x, dw_k, dw_b, xp, xb);

  // 2) cls mean
  k_colmean<<<dim3(Bc, Dc / 256), 256, 0, stream>>>(xp, cls);

  // 3) weight conversion to bf16
  dim3 cvg((WSZ + 255) / 256);
  k_f32_to_bf16<<<cvg, 256, 0, stream>>>(in_w + (size_t)Dc * Dc, wbf_k, WSZ);      // Wk
  k_f32_to_bf16<<<cvg, 256, 0, stream>>>(in_w + (size_t)2 * Dc * Dc, wbf_v, WSZ);  // Wv
  k_f32_to_bf16<<<cvg, 256, 0, stream>>>(xq_w, wbf_xq, WSZ);
  k_f32_to_bf16<<<cvg, 256, 0, stream>>>(xk_w, wbf_xk, WSZ);
  k_f32_to_bf16<<<cvg, 256, 0, stream>>>(yq_w, wbf_yq, WSZ);
  k_f32_to_bf16<<<cvg, 256, 0, stream>>>(yk_w, wbf_yk, WSZ);
  k_f32_to_bf16<<<cvg, 256, 0, stream>>>(v_w, wbf_vf, WSZ);

  // 4) wid_cls / hei_cls, then q projections (fp32, tiny)
  k_small_gemm<<<dim3(Bc, Dc / 64), 64, 0, stream>>>(cls, wid_w, wid_b, whcls, Dc, Dc);
  k_small_gemm<<<dim3(Bc, Dc / 64), 64, 0, stream>>>(cls, hei_w, hei_b, whcls + Bc * Dc, Dc, Dc);
  k_small_gemm<<<dim3(Bc, Dc / 64), 64, 0, stream>>>(whcls, in_w, in_b, qf, Dc, Dc);
  k_small_gemm<<<dim3(Bc, Dc / 64), 64, 0, stream>>>(whcls + Bc * Dc, in_w, in_b, qf + Bc * Dc, Dc, Dc);

  // 5) seven big WMMA GEMMs: (32768x512) @ (512x512)^T
  dim3 gg(Bc * Tc / 128, Dc / 128);
  k_gemm_wmma<<<gg, 256, 0, stream>>>(xb, wbf_k, in_b + Dc, kp, Dc, Dc);
  k_gemm_wmma<<<gg, 256, 0, stream>>>(xb, wbf_v, in_b + 2 * Dc, vmoe, Dc, Dc);
  k_gemm_wmma<<<gg, 256, 0, stream>>>(xb, wbf_xq, xq_b, xqb, Dc, Dc);
  k_gemm_wmma<<<gg, 256, 0, stream>>>(xb, wbf_xk, xk_b, xkb, Dc, Dc);
  k_gemm_wmma<<<gg, 256, 0, stream>>>(xb, wbf_yq, yq_b, yqb, Dc, Dc);
  k_gemm_wmma<<<gg, 256, 0, stream>>>(xb, wbf_yk, yk_b, ykb, Dc, Dc);
  k_gemm_wmma<<<gg, 256, 0, stream>>>(xb, wbf_vf, v_b, vfin, Dc, Dc);

  // 6) MoE attention over T (both wid & hei branches)
  k_scores<<<dim3(2 * Bc * NHEADc, Tc / 256), 256, 0, stream>>>(qf, kp, attn);
  k_softmax4096<<<dim3(2 * Bc * NHEADc), 256, 0, stream>>>(attn);
  k_attn_o<<<dim3(2 * Bc * NHEADc), HDc, 0, stream>>>(attn, vmoe, ovec);

  // 7) temp_w = o @ out_w^T + out_b  (16 rows)
  k_small_gemm<<<dim3(2 * Bc, Dc / 64), 64, 0, stream>>>(ovec, out_w, out_b, tw, Dc, Dc);

  // 8) routing + gaussian gates
  k_router<<<dim3(2 * Bc), 256, 0, stream>>>(tw, rt_w, rt_b, gs_w, gs_b, sgate);

  // 9) gated tensors (alias kp/vmoe memory -- dead after step 6)
  unsigned short* Xw = kp;
  unsigned short* Xh = vmoe;
  k_scale<<<dim3(Bc * Tc), 256, 0, stream>>>(xp, sgate, Xw, Xh);

  // 10) four per-batch 64x64 Grams over K=32768 (WMMA, K-split with atomic reduce)
  dim3 gr(Bc, 16, 8);
  k_gram_wmma<<<gr, 32, 0, stream>>>(Xw, Xw, C4 + 0 * Bc * 4096);
  k_gram_wmma<<<gr, 32, 0, stream>>>(Xh, Xh, C4 + 1 * Bc * 4096);
  k_gram_wmma<<<gr, 32, 0, stream>>>(xqb, xkb, C4 + 2 * Bc * 4096);
  k_gram_wmma<<<gr, 32, 0, stream>>>(yqb, ykb, C4 + 3 * Bc * 4096);

  // 11) priors, softmax, decay gating, M = Sy@Sx
  k_combine<<<dim3(Bc), 64, 0, stream>>>(C4, Mb);

  // 12) out = M @ V (WMMA)
  k_final_wmma<<<dim3(Bc, ROWK / (16 * 8)), 256, 0, stream>>>(Mb, vfin, dout);
}
